// SelfAttentionModel_74457553043541
// MI455X (gfx1250) — compile-verified, matrix operand order checked by reading, not run on hardware
//
#include <hip/hip_runtime.h>
#include <hip/hip_bf16.h>
#include <math.h>

#define BATCH   4
#define SEQ     2048
#define EMBED   512
#define NHEADS  8
#define HD      64
#define FFN     2048
#define MROWS   (BATCH * SEQ)   // 8192 token rows

typedef __attribute__((ext_vector_type(16))) _Float16 v16h;
typedef __attribute__((ext_vector_type(8)))  float    v8f;

union Frag16 { v16h v; int4 q[2]; };

// Load one 16x16x32 WMMA operand fragment from a row-major f16 matrix.
// A-matrix: row = M-row per lane (lane&15). B-matrix: row = N-column per lane
// (source stored column-major / pre-transposed). Per ISA layout: lanes 0-15
// hold K = k0+{0..7,16..23}, lanes 16-31 hold K = k0+{8..15,24..31}
//   ->  exactly two contiguous 16-byte loads (global_load_b128 x2).
__device__ __forceinline__ v16h load_frag_g(const _Float16* __restrict__ base,
                                            int row, int ldk, int k0, int lane) {
  Frag16 f;
  const int kb = k0 + ((lane & 16) ? 8 : 0);
  const _Float16* p = base + (size_t)row * (size_t)ldk + kb;
  f.q[0] = *(const int4*)(p);
  f.q[1] = *(const int4*)(p + 16);
  return f.v;
}

__device__ __forceinline__ v8f wmma_f16(v16h a, v16h b, v8f c) {
  return __builtin_amdgcn_wmma_f32_16x16x32_f16(false, a, false, b,
                                                (short)0, c, false, false);
}

// DPP16 butterfly max all-reduce within each 16-lane group (no LDS traffic,
// unlike __shfl_xor which lowers to ds_bpermute_b32).
template <int CTRL>
__device__ __forceinline__ float dpp_maxstep(float x) {
  const int t = __builtin_amdgcn_update_dpp(0, __float_as_int(x),
                                            CTRL, 0xf, 0xf, true);
  return fmaxf(x, __int_as_float(t));
}
__device__ __forceinline__ float rowmax16(float x) {
  x = dpp_maxstep<0xB1>(x);    // quad_perm [1,0,3,2] : xor 1
  x = dpp_maxstep<0x4E>(x);    // quad_perm [2,3,0,1] : xor 2
  x = dpp_maxstep<0x141>(x);   // row_half_mirror     : combine across 4
  x = dpp_maxstep<0x140>(x);   // row_mirror          : combine across 8
  return x;
}

// ---------------------------------------------------------------------------
// Generic WMMA GEMM:  out = A[M,K] (f16, row-major) * Wt[N,K] (f16, W^T) + bias
// Wave -> 32x64 output slab (2 A-frags x 4 B-frags = 8 wmma / k-step).
// Block = 8 waves stacked in M: covers 256 x 64. grid = (M/256, N/64).
// MODE: 0 = f32 row-major, 1 = f16 row-major, 2 = f16 + exact GELU,
//       3 = f16 [B,H,S,HD] (Q/K; escale folds softmax scale into Q),
//       4 = f16 [B,H,HD,S] (V transposed)
// ---------------------------------------------------------------------------
template <int MODE>
__global__ __launch_bounds__(256)
void gemm_wmma(const _Float16* __restrict__ A, const _Float16* __restrict__ Wt,
               const float* __restrict__ bias, void* __restrict__ out,
               int M, int N, int K, float escale) {
  const int wave = threadIdx.x >> 5;
  const int lane = threadIdx.x & 31;
  const int l15  = lane & 15;
  const int m0 = (blockIdx.x * 8 + wave) * 32;
  const int n0 = blockIdx.y * 64;

  v8f acc[2][4] = {};

  for (int k0 = 0; k0 < K; k0 += 32) {
    const v16h a0 = load_frag_g(A, m0 + l15,      K, k0, lane);
    const v16h a1 = load_frag_g(A, m0 + 16 + l15, K, k0, lane);
#pragma unroll
    for (int t = 0; t < 4; ++t) {
      const v16h b = load_frag_g(Wt, n0 + t * 16 + l15, K, k0, lane);
      acc[0][t] = wmma_f16(a0, b, acc[0][t]);
      acc[1][t] = wmma_f16(a1, b, acc[1][t]);
    }
  }

  // C/D layout: lane holds column n; lanes 0-15 rows +0..7, lanes 16-31 rows +8..15
  const int rb = (lane & 16) ? 8 : 0;
#pragma unroll
  for (int sub = 0; sub < 2; ++sub) {
    const int rbase = m0 + sub * 16 + rb;
#pragma unroll
    for (int t = 0; t < 4; ++t) {
      const int n = n0 + t * 16 + l15;
      const float bb = bias[n];
#pragma unroll
      for (int r = 0; r < 8; ++r) {
        const int m = rbase + r;
        float val = acc[sub][t][r] + bb;
        if (MODE == 0) {
          ((float*)out)[(size_t)m * N + n] = val;
        } else if (MODE == 1) {
          ((_Float16*)out)[(size_t)m * N + n] = (_Float16)val;
        } else if (MODE == 2) {
          val = 0.5f * val * (1.0f + erff(val * 0.70710678118654752f));
          ((_Float16*)out)[(size_t)m * N + n] = (_Float16)val;
        } else if (MODE == 3) {  // [B,H,S,HD], scaled
          const int b = m >> 11, s = m & (SEQ - 1);
          const int h = n >> 6,  d = n & 63;
          ((_Float16*)out)[(((size_t)(b * NHEADS + h) * SEQ) + s) * HD + d] =
              (_Float16)(val * escale);
        } else {                 // [B,H,HD,S]
          const int b = m >> 11, s = m & (SEQ - 1);
          const int h = n >> 6,  d = n & 63;
          ((_Float16*)out)[(((size_t)(b * NHEADS + h) * HD) + d) * SEQ + s] =
              (_Float16)val;
        }
      }
    }
  }
}

// ---------------------------------------------------------------------------
// Flash attention: one wave per (b, h, 16-query tile). Online softmax over
// 32-key chunks. Row-max via DPP butterfly; row-sum via an extra WMMA against
// an all-ones B fragment (result lands directly in the li[] lane layout).
// Scale 1/sqrt(512) is pre-folded into Q by the projection epilogue.
// Q,K: [B,H,S,HD] f16;  V: [B,H,HD,S] f16;  ctx out: [B,S,D] f16 row-major.
// ---------------------------------------------------------------------------
__global__ __launch_bounds__(256)
void attn_wmma(const _Float16* __restrict__ Q, const _Float16* __restrict__ Kh,
               const _Float16* __restrict__ Vt, _Float16* __restrict__ ctx) {
  __shared__ _Float16 Psh[8][16 * 32];   // per-wave 16x32 P staging
  const int wave = threadIdx.x >> 5;
  const int lane = threadIdx.x & 31;
  const int l15  = lane & 15;
  const int gw = blockIdx.x * 8 + wave;  // 4096 waves total
  const int qt = gw & 127;               // 128 query tiles per (b,h)
  const int bh = gw >> 7;                // 0..31
  const int b = bh >> 3, h = bh & 7;

  const _Float16* q = Q  + (size_t)bh * SEQ * HD;
  const _Float16* k = Kh + (size_t)bh * SEQ * HD;
  const _Float16* v = Vt + (size_t)bh * HD * SEQ;
  const int q0 = qt * 16;
  const int rb = (lane & 16) ? 8 : 0;
  _Float16* pm = &Psh[wave][0];

  const v16h aq0 = load_frag_g(q, q0 + l15, HD, 0,  lane);
  const v16h aq1 = load_frag_g(q, q0 + l15, HD, 32, lane);

  v16h ones;
#pragma unroll
  for (int i = 0; i < 16; ++i) ones[i] = (_Float16)1.0f;

  v8f o[4] = {};
  float mi[8], li[8];
#pragma unroll
  for (int r = 0; r < 8; ++r) { mi[r] = -3.0e38f; li[r] = 0.0f; }

  for (int j = 0; j < SEQ; j += 32) {
    // scores: two 16x16 tiles over 32 keys, K-dim = HD = 64 (2 wmma each)
    v8f s0 = {}, s1 = {};
    {
      const v16h b0 = load_frag_g(k, j + l15,      HD, 0,  lane);
      const v16h b1 = load_frag_g(k, j + l15,      HD, 32, lane);
      s0 = wmma_f16(aq0, b0, s0);
      s0 = wmma_f16(aq1, b1, s0);
      const v16h c0 = load_frag_g(k, j + 16 + l15, HD, 0,  lane);
      const v16h c1 = load_frag_g(k, j + 16 + l15, HD, 32, lane);
      s1 = wmma_f16(aq0, c0, s1);
      s1 = wmma_f16(aq1, c1, s1);
    }
    // online softmax per row; max reduced across 16 lanes via DPP
    float alpha[8];
#pragma unroll
    for (int r = 0; r < 8; ++r) {
      const float e0 = s0[r];
      const float e1 = s1[r];
      const float mx = rowmax16(fmaxf(e0, e1));
      const float mn = fmaxf(mi[r], mx);
      alpha[r] = __expf(mi[r] - mn);
      mi[r] = mn;
      const float p0 = __expf(e0 - mn);
      const float p1 = __expf(e1 - mn);
      const int row = rb + r;
      pm[row * 32 + l15]      = (_Float16)p0;   // P[row][key_local]
      pm[row * 32 + 16 + l15] = (_Float16)p1;
#pragma unroll
      for (int t = 0; t < 4; ++t) o[t][r] *= alpha[r];
    }
    // per-wave LDS transpose: DS ops are in-order within a wave; the wait +
    // memory clobber also stops the compiler reordering around the aliasing.
    asm volatile("s_wait_dscnt 0" ::: "memory");
    Frag16 pf;
    {
      const _Float16* ps = pm + l15 * 32 + ((lane & 16) ? 8 : 0);
      pf.q[0] = *(const int4*)(ps);
      pf.q[1] = *(const int4*)(ps + 16);
    }
    // row-sum of P via WMMA with all-ones B: ls[r] = sum_k P[row][k],
    // identical lane layout to li[] -> no cross-lane shuffles needed.
    v8f ls = {};
    ls = wmma_f16(pf.v, ones, ls);
    // PV: K-dim = 32 keys, 4 d-tiles of 16
#pragma unroll
    for (int t = 0; t < 4; ++t) {
      const v16h bv = load_frag_g(v, t * 16 + l15, SEQ, j, lane);
      o[t] = wmma_f16(pf.v, bv, o[t]);
    }
#pragma unroll
    for (int r = 0; r < 8; ++r) li[r] = li[r] * alpha[r] + ls[r];
    asm volatile("s_wait_dscnt 0" ::: "memory");
  }

  // normalize and scatter ctx (f16 row-major [B,S,D])
  _Float16* cptr = ctx + ((size_t)(b * SEQ + q0) * EMBED) + h * HD;
#pragma unroll
  for (int r = 0; r < 8; ++r) {
    const float inv = 1.0f / li[r];
    const int row = rb + r;
#pragma unroll
    for (int t = 0; t < 4; ++t) {
      cptr[(size_t)row * EMBED + t * 16 + l15] = (_Float16)(o[t][r] * inv);
    }
  }
}

// ---------------------------------------------------------------------------
// Fused residual-add + LayerNorm over D=512. One block (256 thr) per row.
// ---------------------------------------------------------------------------
__global__ __launch_bounds__(256)
void add_ln(const float* __restrict__ X, const float* __restrict__ Y,
            const float* __restrict__ gamma, const float* __restrict__ beta,
            float* __restrict__ out32, _Float16* __restrict__ out16) {
  __shared__ float s_sum[256], s_sq[256];
  const int row = blockIdx.x;
  const int t = threadIdx.x;
  const size_t base = (size_t)row * EMBED;
  const float a0 = X[base + t]       + Y[base + t];
  const float a1 = X[base + t + 256] + Y[base + t + 256];
  s_sum[t] = a0 + a1;
  s_sq[t]  = a0 * a0 + a1 * a1;
  __syncthreads();
  for (int off = 128; off > 0; off >>= 1) {
    if (t < off) { s_sum[t] += s_sum[t + off]; s_sq[t] += s_sq[t + off]; }
    __syncthreads();
  }
  const float mean = s_sum[0] * (1.0f / EMBED);
  const float var  = s_sq[0] * (1.0f / EMBED) - mean * mean;
  const float rstd = rsqrtf(var + 1e-5f);
  const float o0 = (a0 - mean) * rstd * gamma[t]       + beta[t];
  const float o1 = (a1 - mean) * rstd * gamma[t + 256] + beta[t + 256];
  out32[base + t]       = o0;
  out32[base + t + 256] = o1;
  if (out16) {
    out16[base + t]       = (_Float16)o0;
    out16[base + t + 256] = (_Float16)o1;
  }
}

__global__ __launch_bounds__(256)
void f32_to_f16(const float* __restrict__ in, _Float16* __restrict__ out, int n) {
  const int i = blockIdx.x * 256 + threadIdx.x;
  if (i < n) out[i] = (_Float16)in[i];
}

// Wt[n][k] = (f16) W[k][n]  (W row-major [K,N]) — B operands column-major.
__global__ __launch_bounds__(256)
void transpose_f16(const float* __restrict__ W, _Float16* __restrict__ Wt,
                   int K, int N) {
  const int i = blockIdx.x * 256 + threadIdx.x;
  if (i < K * N) {
    const int kk = i / N, nn = i % N;
    Wt[(size_t)nn * K + kk] = (_Float16)W[i];
  }
}

extern "C" void kernel_launch(void* const* d_in, const int* in_sizes, int n_in,
                              void* d_out, int out_size, void* d_ws, size_t ws_size,
                              hipStream_t stream) {
  (void)in_sizes; (void)n_in; (void)out_size; (void)ws_size;
  const float* x      = (const float*)d_in[0];
  const float* wq     = (const float*)d_in[1];
  const float* bq     = (const float*)d_in[2];
  const float* wk     = (const float*)d_in[3];
  const float* bk     = (const float*)d_in[4];
  const float* wv     = (const float*)d_in[5];
  const float* bv     = (const float*)d_in[6];
  const float* wo     = (const float*)d_in[7];
  const float* bo     = (const float*)d_in[8];
  const float* w_up   = (const float*)d_in[9];
  const float* b_up   = (const float*)d_in[10];
  const float* w_down = (const float*)d_in[11];
  const float* b_down = (const float*)d_in[12];
  const float* g1     = (const float*)d_in[13];
  const float* beta1  = (const float*)d_in[14];
  const float* g2     = (const float*)d_in[15];
  const float* beta2  = (const float*)d_in[16];

  char* ws = (char*)d_ws;
  size_t off = 0;
  auto alloc = [&](size_t bytes) -> void* {
    void* p = ws + off;
    off = (off + bytes + 255) & ~(size_t)255;
    return p;
  };
  _Float16* xh   = (_Float16*)alloc((size_t)MROWS * EMBED * 2);
  _Float16* qb   = (_Float16*)alloc((size_t)MROWS * EMBED * 2);
  _Float16* kbuf = (_Float16*)alloc((size_t)MROWS * EMBED * 2);
  _Float16* vtb  = (_Float16*)alloc((size_t)MROWS * EMBED * 2);
  _Float16* ctxh = (_Float16*)alloc((size_t)MROWS * EMBED * 2);
  float*    yb   = (float*)   alloc((size_t)MROWS * EMBED * 4);
  float*    x1   = (float*)   alloc((size_t)MROWS * EMBED * 4);
  _Float16* x1h  = (_Float16*)alloc((size_t)MROWS * EMBED * 2);
  _Float16* hid  = (_Float16*)alloc((size_t)MROWS * FFN * 2);
  float*    y2   = (float*)   alloc((size_t)MROWS * EMBED * 4);
  _Float16* wqT  = (_Float16*)alloc((size_t)EMBED * EMBED * 2);
  _Float16* wkT  = (_Float16*)alloc((size_t)EMBED * EMBED * 2);
  _Float16* wvT  = (_Float16*)alloc((size_t)EMBED * EMBED * 2);
  _Float16* woT  = (_Float16*)alloc((size_t)EMBED * EMBED * 2);
  _Float16* wupT = (_Float16*)alloc((size_t)EMBED * FFN * 2);
  _Float16* wdnT = (_Float16*)alloc((size_t)FFN * EMBED * 2);

  // 1) operand conversion / weight transposition (tiny, L2-resident)
  f32_to_f16<<<(MROWS * EMBED) / 256, 256, 0, stream>>>(x, xh, MROWS * EMBED);
  transpose_f16<<<(EMBED * EMBED) / 256, 256, 0, stream>>>(wq, wqT, EMBED, EMBED);
  transpose_f16<<<(EMBED * EMBED) / 256, 256, 0, stream>>>(wk, wkT, EMBED, EMBED);
  transpose_f16<<<(EMBED * EMBED) / 256, 256, 0, stream>>>(wv, wvT, EMBED, EMBED);
  transpose_f16<<<(EMBED * EMBED) / 256, 256, 0, stream>>>(wo, woT, EMBED, EMBED);
  transpose_f16<<<(EMBED * FFN) / 256, 256, 0, stream>>>(w_up, wupT, EMBED, FFN);
  transpose_f16<<<(FFN * EMBED) / 256, 256, 0, stream>>>(w_down, wdnT, FFN, EMBED);

  const dim3 gProj(MROWS / 256, EMBED / 64);   // (32, 8)
  const dim3 gUp  (MROWS / 256, FFN / 64);     // (32, 32)
  const float qscale = 0.04419417382415922f;   // 1/sqrt(EMBED) per reference

  // 2) QKV projections (WMMA), head-layout epilogues, scale folded into Q
  gemm_wmma<3><<<gProj, 256, 0, stream>>>(xh, wqT, bq, qb,   MROWS, EMBED, EMBED, qscale);
  gemm_wmma<3><<<gProj, 256, 0, stream>>>(xh, wkT, bk, kbuf, MROWS, EMBED, EMBED, 1.0f);
  gemm_wmma<4><<<gProj, 256, 0, stream>>>(xh, wvT, bv, vtb,  MROWS, EMBED, EMBED, 1.0f);

  // 3) flash attention (WMMA QK^T + row-sum + PV)
  attn_wmma<<<(BATCH * NHEADS * (SEQ / 16)) / 8, 256, 0, stream>>>(qb, kbuf, vtb, ctxh);

  // 4) output projection + residual LN1
  gemm_wmma<0><<<gProj, 256, 0, stream>>>(ctxh, woT, bo, yb, MROWS, EMBED, EMBED, 1.0f);
  add_ln<<<MROWS, 256, 0, stream>>>(x, yb, g1, beta1, x1, x1h);

  // 5) FFN: up + exact GELU (fused), down, residual LN2 -> d_out
  gemm_wmma<2><<<gUp,   256, 0, stream>>>(x1h, wupT, b_up,   hid, MROWS, FFN,   EMBED, 1.0f);
  gemm_wmma<0><<<gProj, 256, 0, stream>>>(hid, wdnT, b_down, y2,  MROWS, EMBED, FFN, 1.0f);
  add_ln<<<MROWS, 256, 0, stream>>>(x1, y2, g2, beta2, (float*)d_out, (_Float16*)nullptr);
}
